// LCGN_60035052863771
// MI455X (gfx1250) — matrix operand order repaired and checked by reading, not compiled, not run on hardware
//
#include <hip/hip_runtime.h>
#include <hip/hip_bf16.h>
#include <math.h>

typedef __attribute__((ext_vector_type(16))) __bf16 v16bf;
typedef __attribute__((ext_vector_type(8)))  __bf16 v8bf;
typedef __attribute__((ext_vector_type(2)))  __bf16 v2bf;
typedef __attribute__((ext_vector_type(8)))  float  v8f;
typedef __attribute__((ext_vector_type(4)))  float  v4f;

constexpr int BB = 128, NN_ = 196, SS = 32, DFEAT = 1024, CTX = 512, CMD = 512, TT = 4;
constexpr float NEGBIG = -1e30f;

union BfPack { v16bf v; v8bf h[2]; };

// ---------------------------------------------------------------------------
// Tiled bf16 WMMA GEMM: C[M,Nn] (f32) = epilogue( A[M,K] @ B + bias )
//   MODE 0: C = r (+bias if bias)         MODE 1: C = elu(r + bias)
//   MODE 2: C += r                        MODE 3: C *= (r + bias)
//   MODE 4: C = (col < keycnt[bz]) ? alpha*r : NEG   (attention scores)
// B_IS_KN: true -> B stored [K,Nn] row-major; false -> B stored [Nn,K] (NT)
// Batched via blockIdx.z (strides sA/sB/sC). 256 thr = 8 waves; 128x128 C tile;
// wave = 32x64 sub-tile = 2x4 v_wmma_f32_16x16x32_bf16 per 32-deep K step.
// Register double-buffered global->LDS pipeline; bf16 stored transposed in LDS
// so every fragment read is contiguous ds_load_b128 pairs.
// ---------------------------------------------------------------------------
template<int MODE, bool B_IS_KN>
__global__ __launch_bounds__(256)
void gemm_wmma_bf16(const float* __restrict__ A, const float* __restrict__ Bm,
                    const float* __restrict__ bias, float* __restrict__ C,
                    int M, int Nn, int K,
                    long lda, long ldb, long ldc,
                    long sA, long sB, long sC,
                    const float* __restrict__ a_rowscale,
                    float alpha, const int* __restrict__ keycnt)
{
  __shared__ __align__(16) __bf16 lA[128 * 32];
  __shared__ __align__(16) __bf16 lB[128 * 32];

  const int tid   = threadIdx.x;
  const int lane  = tid & 31;
  const int wave  = tid >> 5;
  const int waveM = wave & 3;   // 4 waves along M, 32 rows each
  const int waveN = wave >> 2;  // 2 waves along N, 64 cols each
  const int bz    = blockIdx.z;
  const int tileM = blockIdx.y * 128;
  const int tileN = blockIdx.x * 128;
  const float* Ab = A  + (long)bz * sA;
  const float* Bb = Bm + (long)bz * sB;

  const bool rowFullA = (tileM + 128 <= M);
  const bool colFullB = (tileN + 128 <= Nn);

  v8f acc[2][4];
  #pragma unroll
  for (int i = 0; i < 2; ++i)
    #pragma unroll
    for (int j = 0; j < 4; ++j)
      #pragma unroll
      for (int e = 0; e < 8; ++e) acc[i][j][e] = 0.0f;

  // A / NT-B loader mapping: one row-segment of 16 consecutive k per thread
  const int arow = tid >> 1;          // 0..127
  const int aseg = (tid & 1) * 16;    // 0 or 16
  // KN-B loader mapping: two k rows x 8 n per thread (packed bf16-pair stores)
  const int bk0  = (tid & 15) * 2;    // even k in 0..30
  const int bn0  = (tid >> 4) * 8;    // 0..120

  float ra[16];   // A stage regs
  float rb[16];   // B stage regs (KN: [0..7]=row bk0, [8..15]=row bk0+1)

  auto loadTiles = [&](int k0) {
    // ---- A (and NT-B shares the pattern)
    {
      const int grow = tileM + arow;
      const long rowi = (grow < M) ? grow : (M - 1);
      const float* src = Ab + rowi * lda + k0 + aseg;
      if (rowFullA && (k0 + 32 <= K)) {
        const v4f* s4 = (const v4f*)src;
        #pragma unroll
        for (int c = 0; c < 4; ++c) {
          v4f q = s4[c];
          #pragma unroll
          for (int e = 0; e < 4; ++e) ra[c * 4 + e] = q[e];
        }
      } else {
        #pragma unroll
        for (int j = 0; j < 16; ++j) {
          int kk = k0 + aseg + j;
          int kc = (kk < K) ? kk : (K - 1);
          float f = Ab[rowi * lda + kc];               // always in-bounds
          ra[j] = (grow < M && kk < K) ? f : 0.0f;     // v_cndmask, no branch
        }
      }
      if (a_rowscale) {
        float rs = a_rowscale[rowi];
        #pragma unroll
        for (int j = 0; j < 16; ++j) ra[j] *= rs;
      }
    }
    // ---- B
    if (B_IS_KN) {
      const int kk = k0 + bk0;
      const bool k0ok = kk < K, k1ok = (kk + 1) < K;
      const long kr0 = k0ok ? kk : (K - 1);
      const long kr1 = k1ok ? (kk + 1) : (K - 1);
      const float* s0 = Bb + kr0 * ldb + tileN + bn0;
      const float* s1 = Bb + kr1 * ldb + tileN + bn0;
      if (colFullB && (k0 + 32 <= K)) {
        const v4f* p0 = (const v4f*)s0;
        const v4f* p1 = (const v4f*)s1;
        #pragma unroll
        for (int c = 0; c < 2; ++c) {
          v4f q0 = p0[c], q1 = p1[c];
          #pragma unroll
          for (int e = 0; e < 4; ++e) { rb[c * 4 + e] = q0[e]; rb[8 + c * 4 + e] = q1[e]; }
        }
      } else {
        #pragma unroll
        for (int j = 0; j < 8; ++j) {
          int gn = tileN + bn0 + j;
          long gc = (gn < Nn) ? gn : (Nn - 1);
          float f0 = Bb[kr0 * ldb + gc];
          float f1 = Bb[kr1 * ldb + gc];
          rb[j]     = (gn < Nn && k0ok) ? f0 : 0.0f;
          rb[8 + j] = (gn < Nn && k1ok) ? f1 : 0.0f;
        }
      }
    } else {
      const int gn = tileN + arow;
      const long rowi = (gn < Nn) ? gn : (Nn - 1);
      const float* src = Bb + rowi * ldb + k0 + aseg;
      if (colFullB && (k0 + 32 <= K)) {
        const v4f* s4 = (const v4f*)src;
        #pragma unroll
        for (int c = 0; c < 4; ++c) {
          v4f q = s4[c];
          #pragma unroll
          for (int e = 0; e < 4; ++e) rb[c * 4 + e] = q[e];
        }
      } else {
        #pragma unroll
        for (int j = 0; j < 16; ++j) {
          int kk = k0 + aseg + j;
          int kc = (kk < K) ? kk : (K - 1);
          float f = Bb[rowi * ldb + kc];
          rb[j] = (gn < Nn && kk < K) ? f : 0.0f;
        }
      }
    }
  };

  auto storeLds = [&]() {
    v8bf pa0, pa1;
    #pragma unroll
    for (int e = 0; e < 8; ++e) { pa0[e] = (__bf16)ra[e]; pa1[e] = (__bf16)ra[8 + e]; }
    *reinterpret_cast<v8bf*>(&lA[arow * 32 + aseg])     = pa0;
    *reinterpret_cast<v8bf*>(&lA[arow * 32 + aseg + 8]) = pa1;
    if (B_IS_KN) {
      #pragma unroll
      for (int j = 0; j < 8; ++j) {
        v2bf p; p[0] = (__bf16)rb[j]; p[1] = (__bf16)rb[8 + j];
        *reinterpret_cast<v2bf*>(&lB[(bn0 + j) * 32 + bk0]) = p;
      }
    } else {
      v8bf pb0, pb1;
      #pragma unroll
      for (int e = 0; e < 8; ++e) { pb0[e] = (__bf16)rb[e]; pb1[e] = (__bf16)rb[8 + e]; }
      *reinterpret_cast<v8bf*>(&lB[arow * 32 + aseg])     = pb0;
      *reinterpret_cast<v8bf*>(&lB[arow * 32 + aseg + 8]) = pb1;
    }
  };

  loadTiles(0);
  for (int k0 = 0; k0 < K; k0 += 32) {
    storeLds();
    __syncthreads();
    if (k0 + 32 < K) loadTiles(k0 + 32);   // global loads overlap WMMA below

    const int lm  = lane & 15;
    const int akb = (lane < 16) ? 0 : 8;
    const int bkg = (lane < 16) ? 0 : 8;
    v16bf afr[2], bfr[4];
    #pragma unroll
    for (int mt = 0; mt < 2; ++mt) {
      int row = waveM * 32 + mt * 16 + lm;
      BfPack p;
      p.h[0] = *reinterpret_cast<const v8bf*>(&lA[row * 32 + akb]);
      p.h[1] = *reinterpret_cast<const v8bf*>(&lA[row * 32 + akb + 16]);
      afr[mt] = p.v;
    }
    #pragma unroll
    for (int nt = 0; nt < 4; ++nt) {
      int col = waveN * 64 + nt * 16 + lm;
      BfPack p;
      p.h[0] = *reinterpret_cast<const v8bf*>(&lB[col * 32 + bkg]);
      p.h[1] = *reinterpret_cast<const v8bf*>(&lB[col * 32 + bkg + 16]);
      bfr[nt] = p.v;
    }
    #pragma unroll
    for (int mt = 0; mt < 2; ++mt)
      #pragma unroll
      for (int nt = 0; nt < 4; ++nt)
        acc[mt][nt] = __builtin_amdgcn_wmma_f32_16x16x32_bf16(
            false, afr[mt], false, bfr[nt], (short)0, acc[mt][nt], false, false);
    __syncthreads();
  }

  // ---- epilogue (C/D layout: VGPR r -> M = r + 8*(lane>=16), N = lane%16)
  float* Cb = C + (long)bz * sC;
  const int lm = lane & 15;
  const int mo = (lane >> 4) * 8;
  int lim = 0;
  if (MODE == 4) lim = keycnt[bz];

  auto epi = [&](int row, int col, float v) {
    long idx = (long)row * ldc + col;
    if (MODE == 0)      { if (bias) v += bias[col]; Cb[idx] = v; }
    else if (MODE == 1) { v += bias[col]; Cb[idx] = v > 0.f ? v : (__expf(v) - 1.f); }
    else if (MODE == 2) { Cb[idx] += v; }
    else if (MODE == 3) { v += bias[col]; Cb[idx] *= v; }
    else                { float s = v * alpha; Cb[idx] = (col < lim) ? s : NEGBIG; }
  };

  if (rowFullA && colFullB) {
    #pragma unroll
    for (int mt = 0; mt < 2; ++mt)
      #pragma unroll
      for (int nt = 0; nt < 4; ++nt)
        #pragma unroll
        for (int r = 0; r < 8; ++r)
          epi(tileM + waveM * 32 + mt * 16 + mo + r,
              tileN + waveN * 64 + nt * 16 + lm, acc[mt][nt][r]);
  } else {
    #pragma unroll
    for (int mt = 0; mt < 2; ++mt)
      #pragma unroll
      for (int nt = 0; nt < 4; ++nt)
        #pragma unroll
        for (int r = 0; r < 8; ++r) {
          int row = tileM + waveM * 32 + mt * 16 + mo + r;
          int col = tileN + waveN * 64 + nt * 16 + lm;
          if (row < M && col < Nn) epi(row, col, acc[mt][nt][r]);
        }
  }
}

// ---------------------------------------------------------------------------
// Small helper kernels
// ---------------------------------------------------------------------------
__global__ void row_l2scale(const float* __restrict__ x, float* __restrict__ scale,
                            int rows, int cols) {
  int row = blockIdx.x, tid = threadIdx.x;
  __shared__ float red[256];
  float s = 0.f;
  for (int c = tid; c < cols; c += 256) { float v = x[(long)row * cols + c]; s += v * v; }
  red[tid] = s; __syncthreads();
  for (int k = 128; k > 0; k >>= 1) { if (tid < k) red[tid] += red[tid + k]; __syncthreads(); }
  if (tid == 0) scale[row] = 1.0f / fmaxf(sqrtf(red[0]), 1e-12f);
}

__global__ void bcast_rows(float* __restrict__ x, const float* __restrict__ m, long tot) {
  long i = (long)blockIdx.x * blockDim.x + threadIdx.x;
  if (i < tot) x[i] = m[i & (CTX - 1)];
}

__global__ void rowmul(float* __restrict__ x, const float* __restrict__ vec) {
  long i = (long)blockIdx.x * blockDim.x + threadIdx.x;
  const long tot = (long)BB * NN_ * CTX;
  if (i < tot) {
    int col = (int)(i & (CTX - 1));
    int bn  = (int)(i >> 9);        // / 512
    int b   = bn / NN_;
    x[i] *= vec[(long)b * CTX + col];
  }
}

__global__ void softmax_rows(float* __restrict__ x, int cols) {
  int row = blockIdx.x, tid = threadIdx.x;
  __shared__ float red[256];
  float m = -INFINITY;
  for (int c = tid; c < cols; c += 256) m = fmaxf(m, x[(long)row * cols + c]);
  red[tid] = m; __syncthreads();
  for (int k = 128; k > 0; k >>= 1) { if (tid < k) red[tid] = fmaxf(red[tid], red[tid + k]); __syncthreads(); }
  m = red[0]; __syncthreads();
  float s = 0.f;
  for (int c = tid; c < cols; c += 256) {
    float e = __expf(x[(long)row * cols + c] - m);
    x[(long)row * cols + c] = e; s += e;
  }
  red[tid] = s; __syncthreads();
  for (int k = 128; k > 0; k >>= 1) { if (tid < k) red[tid] += red[tid + k]; __syncthreads(); }
  float inv = 1.0f / red[0];
  for (int c = tid; c < cols; c += 256) x[(long)row * cols + c] *= inv;
}

// textual attention: raw[s]=sum_d qcmd[d]*lstm[b,s,d]*w2l[d]+b2l; masked softmax; cmd=att@lstm
__global__ void textual_cmd(const float* __restrict__ q_cmd, const float* __restrict__ lstm,
                            const float* __restrict__ w2l, const float* __restrict__ b2l,
                            const int* __restrict__ q_len, float* __restrict__ cmd) {
  int b = blockIdx.x, tid = threadIdx.x;
  __shared__ float raw[SS];
  __shared__ float att[SS];
  int s = tid >> 3, dlane = tid & 7;
  const float* qc = q_cmd + (long)b * CMD;
  const float* lr = lstm + ((long)b * SS + s) * CMD;
  float part = 0.f;
  for (int d = dlane; d < CMD; d += 8) part += qc[d] * lr[d] * w2l[d];
  for (int off = 4; off > 0; off >>= 1) part += __shfl_down(part, off, 8);
  if (dlane == 0) {
    float r = part + b2l[0];
    raw[s] = (s < q_len[b]) ? r : NEGBIG;
  }
  __syncthreads();
  if (tid == 0) {
    float m = -INFINITY;
    for (int i = 0; i < SS; ++i) m = fmaxf(m, raw[i]);
    float sum = 0.f;
    for (int i = 0; i < SS; ++i) { att[i] = __expf(raw[i] - m); sum += att[i]; }
    float inv = 1.0f / sum;
    for (int i = 0; i < SS; ++i) att[i] *= inv;
  }
  __syncthreads();
  for (int d = tid; d < CMD; d += 256) {
    float acc = 0.f;
    for (int i = 0; i < SS; ++i) acc += att[i] * lstm[((long)b * SS + i) * CMD + d];
    cmd[(long)b * CMD + d] = acc;
  }
}

// ---------------------------------------------------------------------------
extern "C" void kernel_launch(void* const* d_in, const int* in_sizes, int n_in,
                              void* d_out, int out_size, void* d_ws, size_t ws_size,
                              hipStream_t stream) {
  (void)in_sizes; (void)n_in; (void)out_size; (void)ws_size;
  const float* images   = (const float*)d_in[0];
  const float* q_enc    = (const float*)d_in[1];
  const float* lstm     = (const float*)d_in[2];
  const int*   q_len    = (const int*)d_in[3];
  const int*   ent_num  = (const int*)d_in[4];
  const float* initKB_W = (const float*)d_in[5];
  const float* initKB_b = (const float*)d_in[6];
  const float* initMem  = (const float*)d_in[7];
  const float* qIn_W    = (const float*)d_in[8];
  const float* qIn_b    = (const float*)d_in[9];
  const float* Wt       = (const float*)d_in[10];
  const float* bt       = (const float*)d_in[11];
  const float* c2l_W    = (const float*)d_in[12];
  const float* c2l_b    = (const float*)d_in[13];
  const float* pl_W     = (const float*)d_in[14];
  const float* pl_b     = (const float*)d_in[15];
  const float* pc_W     = (const float*)d_in[16];
  const float* pc_b     = (const float*)d_in[17];
  const float* qW       = (const float*)d_in[18];
  const float* qb       = (const float*)d_in[19];
  const float* kW       = (const float*)d_in[20];
  const float* kb_      = (const float*)d_in[21];
  const float* vW       = (const float*)d_in[22];
  const float* vb       = (const float*)d_in[23];
  const float* pkW      = (const float*)d_in[24];
  const float* pkb      = (const float*)d_in[25];
  const float* pvW      = (const float*)d_in[26];
  const float* pvb      = (const float*)d_in[27];
  const float* muW      = (const float*)d_in[28];
  const float* mub      = (const float*)d_in[29];
  const float* cbW      = (const float*)d_in[30];
  const float* cbb      = (const float*)d_in[31];

  float* ws = (float*)d_ws;
  const long P = (long)BB * NN_ * CTX;    // 12,845,056 floats per activation
  float* x_loc = ws;
  float* ctxA  = ws + P;
  float* ctxB  = ws + 2 * P;
  float* pp    = ws + 3 * P;              // pl*pc, later reused for message
  float* qbuf  = ws + 4 * P;
  float* kbuf  = ws + 5 * P;
  float* vbuf  = ws + 6 * P;
  float* score = ws + 7 * P;              // [B,196,196]
  float* sm    = score + (long)BB * NN_ * NN_;
  float* rscale = sm;  sm += (long)BB * NN_;
  float* qihid  = sm;  sm += BB * CMD;
  float* qcmd   = sm;  sm += BB * CMD;
  float* cmdv   = sm;  sm += BB * CMD;
  float* pkv    = sm;  sm += BB * CTX;
  float* pvv    = sm;  sm += BB * CTX;

  const int Mbig = BB * NN_;              // 25088
  const dim3 blk(256);
  const float inv_sqrt_d = 0.044194173824159216f;  // 1/sqrt(512)

  // STEM normalize (per-row L2 scale) fused into first GEMM via a_rowscale
  row_l2scale<<<Mbig, blk, 0, stream>>>(images, rscale, Mbig, DFEAT);
  {
    dim3 g(CTX / 128, (Mbig + 127) / 128, 1);
    gemm_wmma_bf16<0, true><<<g, blk, 0, stream>>>(images, initKB_W, initKB_b, x_loc,
        Mbig, CTX, DFEAT, DFEAT, CTX, CTX, 0, 0, 0, rscale, 1.f, nullptr);
  }
  bcast_rows<<<(int)((P + 255) / 256), blk, 0, stream>>>(ctxA, initMem, P);
  {
    dim3 g(CMD / 128, 1, 1);
    gemm_wmma_bf16<1, true><<<g, blk, 0, stream>>>(q_enc, qIn_W, qIn_b, qihid,
        BB, CMD, CMD, CMD, CMD, CMD, 0, 0, 0, nullptr, 1.f, nullptr);
  }

  float* cur = ctxA; float* alt = ctxB;
  const dim3 gB(CTX / 128, (Mbig + 127) / 128, 1);
  const dim3 gS(CMD / 128, 1, 1);
  const int nElemBlocks = (int)(((long)Mbig * CTX + 255) / 256);

  for (int t = 0; t < TT; ++t) {
    // q_cmd = qi_hidden @ Wt[t] + bt[t]
    gemm_wmma_bf16<0, true><<<gS, blk, 0, stream>>>(qihid, Wt + (long)t * CMD * CMD,
        bt + (long)t * CMD, qcmd, BB, CMD, CMD, CMD, CMD, CMD, 0, 0, 0, nullptr, 1.f, nullptr);
    textual_cmd<<<BB, blk, 0, stream>>>(qcmd, lstm, c2l_W, c2l_b, q_len, cmdv);
    gemm_wmma_bf16<0, true><<<gS, blk, 0, stream>>>(cmdv, pkW, pkb, pkv,
        BB, CTX, CMD, CMD, CTX, CTX, 0, 0, 0, nullptr, 1.f, nullptr);
    gemm_wmma_bf16<0, true><<<gS, blk, 0, stream>>>(cmdv, pvW, pvb, pvv,
        BB, CTX, CMD, CMD, CTX, CTX, 0, 0, 0, nullptr, 1.f, nullptr);

    // pp = (x_loc@pl_W + b) * (x_ctx@pc_W + b)   (fused multiply-into epilogue)
    gemm_wmma_bf16<0, true><<<gB, blk, 0, stream>>>(x_loc, pl_W, pl_b, pp,
        Mbig, CTX, CTX, CTX, CTX, CTX, 0, 0, 0, nullptr, 1.f, nullptr);
    gemm_wmma_bf16<3, true><<<gB, blk, 0, stream>>>(cur, pc_W, pc_b, pp,
        Mbig, CTX, CTX, CTX, CTX, CTX, 0, 0, 0, nullptr, 1.f, nullptr);

    // q/k/v = [x_loc, x_ctx, pp] @ W + b, split into three K=512 chunks
    const float* srcs[3] = { x_loc, cur, pp };
    float*       dsts[3] = { qbuf, kbuf, vbuf };
    const float* Ws[3]   = { qW, kW, vW };
    const float* bs[3]   = { qb, kb_, vb };
    for (int o = 0; o < 3; ++o) {
      gemm_wmma_bf16<0, true><<<gB, blk, 0, stream>>>(srcs[0], Ws[o], bs[o], dsts[o],
          Mbig, CTX, CTX, CTX, CTX, CTX, 0, 0, 0, nullptr, 1.f, nullptr);
      gemm_wmma_bf16<2, true><<<gB, blk, 0, stream>>>(srcs[1], Ws[o] + (long)CTX * CTX, nullptr,
          dsts[o], Mbig, CTX, CTX, CTX, CTX, CTX, 0, 0, 0, nullptr, 1.f, nullptr);
      gemm_wmma_bf16<2, true><<<gB, blk, 0, stream>>>(srcs[2], Ws[o] + 2l * CTX * CTX, nullptr,
          dsts[o], Mbig, CTX, CTX, CTX, CTX, CTX, 0, 0, 0, nullptr, 1.f, nullptr);
    }
    rowmul<<<nElemBlocks, blk, 0, stream>>>(kbuf, pkv);
    rowmul<<<nElemBlocks, blk, 0, stream>>>(vbuf, pvv);

    // score = mask(q @ k^T * inv_sqrt_d)  (batched NT gemm + key-mask epilogue)
    {
      dim3 g((NN_ + 127) / 128, (NN_ + 127) / 128, BB);
      gemm_wmma_bf16<4, false><<<g, blk, 0, stream>>>(qbuf, kbuf, nullptr, score,
          NN_, NN_, CTX, CTX, CTX, NN_,
          (long)NN_ * CTX, (long)NN_ * CTX, (long)NN_ * NN_,
          nullptr, inv_sqrt_d, ent_num);
    }
    softmax_rows<<<BB * NN_, blk, 0, stream>>>(score, NN_);

    // message = prob @ v  (batched NN gemm, K=196 guarded) -> reuse pp
    {
      dim3 g(CTX / 128, (NN_ + 127) / 128, BB);
      gemm_wmma_bf16<0, true><<<g, blk, 0, stream>>>(score, vbuf, nullptr, pp,
          NN_, CTX, NN_, NN_, CTX, CTX,
          (long)NN_ * NN_, (long)NN_ * CTX, (long)NN_ * CTX,
          nullptr, 1.f, nullptr);
    }

    // x_ctx_new = [x_ctx, message] @ muW + b
    gemm_wmma_bf16<0, true><<<gB, blk, 0, stream>>>(cur, muW, mub, alt,
        Mbig, CTX, CTX, CTX, CTX, CTX, 0, 0, 0, nullptr, 1.f, nullptr);
    gemm_wmma_bf16<2, true><<<gB, blk, 0, stream>>>(pp, muW + (long)CTX * CTX, nullptr, alt,
        Mbig, CTX, CTX, CTX, CTX, CTX, 0, 0, 0, nullptr, 1.f, nullptr);
    float* tmp = cur; cur = alt; alt = tmp;
  }

  // out = [x_loc, x_ctx] @ cbW + b
  float* outp = (float*)d_out;
  gemm_wmma_bf16<0, true><<<gB, blk, 0, stream>>>(x_loc, cbW, cbb, outp,
      Mbig, CTX, CTX, CTX, CTX, CTX, 0, 0, 0, nullptr, 1.f, nullptr);
  gemm_wmma_bf16<2, true><<<gB, blk, 0, stream>>>(cur, cbW + (long)CTX * CTX, nullptr, outp,
      Mbig, CTX, CTX, CTX, CTX, CTX, 0, 0, 0, nullptr, 1.f, nullptr);
}
